// GTrans_block_14388140442031
// MI455X (gfx1250) — compile-verified, matrix operand order checked by reading, not run on hardware
//
#include <hip/hip_runtime.h>
#include <hip/hip_bf16.h>
#include <math.h>

// ---------------- problem constants ----------------
#define NN      25000
#define EEDGE   400000
#define BG      32
#define NODE_IN 128
#define EDGE_IN 64
#define NODE_OUT 64
#define EDGE_OUT 64
#define GLOB_IN 64
#define GLOB_OUT 64
#define NH      3
#define HC      (NH * NODE_OUT)   // 192

typedef __attribute__((ext_vector_type(16))) _Float16 v16h;
typedef __attribute__((ext_vector_type(8)))  float    v8f;

union AFrag { v16h v; _Float16 h[16]; unsigned u[8]; };
union CFrag { v8f  v; float     f[8];  };

// K-offset (within a 32-wide K chunk) of element e (0..15) of an A/B fragment
// for this lane, per CDNA5 ISA 7.12.2 (16-bit A 16x32 layout; B mirrored).
__device__ __forceinline__ int wmma_koff(int lane, int e) {
    int vg = e >> 1, lo = e & 1;
    int base = (vg < 4) ? (2 * vg) : (16 + 2 * (vg - 4));
    return base + ((lane & 16) ? 8 : 0) + lo;
}

// A fragment from an f32 row (unconditional, vectorized float2 + cvt_pk)
__device__ __forceinline__ void load_afrag_f32(const float* __restrict__ row, int kk, int lane, AFrag& a) {
    const float* r = row + kk + ((lane & 16) ? 8 : 0);
#pragma unroll
    for (int p = 0; p < 8; ++p) {
        int kb = (p < 4) ? 2 * p : 16 + 2 * (p - 4);
        float2 v = *(const float2*)(r + kb);
        a.h[2 * p]     = (_Float16)v.x;
        a.h[2 * p + 1] = (_Float16)v.y;
    }
}

// A fragment from an f16 row (8 dword loads)
__device__ __forceinline__ void load_afrag_f16(const _Float16* __restrict__ row, int kk, int lane, AFrag& a) {
    const _Float16* r = row + kk + ((lane & 16) ? 8 : 0);
#pragma unroll
    for (int p = 0; p < 8; ++p) {
        int kb = (p < 4) ? 2 * p : 16 + 2 * (p - 4);
        a.u[p] = *(const unsigned*)(r + kb);
    }
}

// B fragment from pre-swizzled packed f16 weights: one contiguous 32B load.
// Packed layout: halves[((tn*chunks + kc)*32 + lane)*16 + e]
__device__ __forceinline__ v16h load_bfrag_packed(const _Float16* __restrict__ Wp,
                                                  int tn, int kc, int chunks, int lane) {
    return *(const v16h*)(Wp + ((((size_t)tn * chunks + kc) * 32) + lane) * 16);
}

__device__ __forceinline__ unsigned f2ord(float f) {
    unsigned u = __float_as_uint(f);
    return (u & 0x80000000u) ? ~u : (u | 0x80000000u);
}
__device__ __forceinline__ float ord2f(unsigned o) {
    unsigned u = (o & 0x80000000u) ? (o & 0x7FFFFFFFu) : ~o;
    return __uint_as_float(u);
}

// ---------------- init / conversion / packing -------------------------------
__global__ void zero_kernel(float* p, long n) {
    long i = (long)blockIdx.x * blockDim.x + threadIdx.x;
    long stride = (long)gridDim.x * blockDim.x;
    for (; i < n; i += stride) p[i] = 0.0f;
}

__global__ void cvt_f16_kernel(const float* __restrict__ in, _Float16* __restrict__ out, long n) {
    long i = (long)blockIdx.x * blockDim.x + threadIdx.x;
    if (i < n) out[i] = (_Float16)in[i];
}

// Pre-swizzle a row-major f32 weight [K x Ncols] into B-fragment order (f16).
__global__ void pack_weight_kernel(const float* __restrict__ W, _Float16* __restrict__ out,
                                   int K, int Ncols) {
    int i = blockIdx.x * blockDim.x + threadIdx.x;
    if (i >= K * Ncols) return;
    int chunks = K >> 5;
    int e    = i & 15;
    int lane = (i >> 4) & 31;
    int kc   = (i >> 9) % chunks;
    int tn   = (i >> 9) / chunks;
    int k    = kc * 32 + wmma_koff(lane, e);
    int col  = tn * 16 + (lane & 15);
    out[i] = (_Float16)W[(size_t)k * Ncols + col];
}

// ---------------- generic WMMA GEMM (f16 A, packed f16 W) -------------------
__global__ void wmma_gemm_f16(const _Float16* __restrict__ A, const _Float16* __restrict__ Wp,
                              const float* __restrict__ bias, float* __restrict__ D,
                              int M, int K, int Ncols, int doRelu)
{
    int lane = threadIdx.x & 31;
    int wave = blockIdx.x * (blockDim.x >> 5) + (threadIdx.x >> 5);
    int tilesN = Ncols >> 4;
    int tilesM = (M + 15) >> 4;
    if (wave >= tilesM * tilesN) return;
    int tm = wave / tilesN, tn = wave % tilesN;
    int chunks = K >> 5;

    int arow = tm * 16 + (lane & 15);
    int bcol = tn * 16 + (lane & 15);
    int rclamp = arow < M ? arow : M - 1;           // clamp; stores are guarded
    const _Float16* Arow = A + (size_t)rclamp * K;

    CFrag acc;
#pragma unroll
    for (int i = 0; i < 8; ++i) acc.f[i] = 0.0f;

    for (int kc = 0; kc < chunks; ++kc) {
        AFrag a;
        load_afrag_f16(Arow, kc * 32, lane, a);
        v16h b = load_bfrag_packed(Wp, tn, kc, chunks, lane);
        acc.v = __builtin_amdgcn_wmma_f32_16x16x32_f16(
            false, a.v, false, b, (short)0, acc.v, false, false);
    }

    float bv = bias[bcol];
#pragma unroll
    for (int e = 0; e < 8; ++e) {
        int row = tm * 16 + e + ((lane & 16) ? 8 : 0);
        if (row < M) {
            float v = acc.f[e] + bv;
            if (doRelu) v = fmaxf(v, 0.0f);
            D[(size_t)row * Ncols + bcol] = v;
        }
    }
}

// ---------------- edge MLP: e = relu(cat(x[src], edge_attr) @ We + b) -------
__global__ void edge_mlp_kernel(const _Float16* __restrict__ xh, const float* __restrict__ ea,
                                const int* __restrict__ ei, const _Float16* __restrict__ pWe,
                                const float* __restrict__ bias, float* __restrict__ eout,
                                _Float16* __restrict__ eh)
{
    int lane = threadIdx.x & 31;
    int wave = blockIdx.x * (blockDim.x >> 5) + (threadIdx.x >> 5);
    const int tilesN = EDGE_OUT >> 4;               // 4
    const int chunks = (NODE_IN + EDGE_IN) >> 5;    // 6
    if (wave >= (EEDGE / 16) * tilesN) return;
    int tm = wave / tilesN, tn = wave % tilesN;

    int arow = tm * 16 + (lane & 15);               // always < E
    int bcol = tn * 16 + (lane & 15);
    int src  = ei[arow];
    const _Float16* xr = xh + (size_t)src * NODE_IN;
    const float*    er = ea + (size_t)arow * EDGE_IN;

    CFrag acc;
#pragma unroll
    for (int i = 0; i < 8; ++i) acc.f[i] = 0.0f;

#pragma unroll
    for (int kc = 0; kc < chunks; ++kc) {
        AFrag a;
        int kk = kc * 32;
        if (kk < NODE_IN) load_afrag_f16(xr, kk, lane, a);          // uniform per chunk
        else              load_afrag_f32(er, kk - NODE_IN, lane, a);
        v16h b = load_bfrag_packed(pWe, tn, kc, chunks, lane);
        acc.v = __builtin_amdgcn_wmma_f32_16x16x32_f16(
            false, a.v, false, b, (short)0, acc.v, false, false);
    }

    float bv = bias[bcol];
#pragma unroll
    for (int e = 0; e < 8; ++e) {
        int row = tm * 16 + e + ((lane & 16) ? 8 : 0);
        float v = fmaxf(acc.f[e] + bv, 0.0f);
        eout[(size_t)row * EDGE_OUT + bcol] = v;
        eh[(size_t)row * EDGE_OUT + bcol]   = (_Float16)v;
    }
}

// ---------------- fused ee-tile compute (WMMA) into LDS ---------------------
// ee[16x192] = e_tile[16x64] @ Wed[64x192], one 16-edge tile per wave.
__device__ __forceinline__ void compute_ee_tile(const _Float16* __restrict__ eh,
                                                const _Float16* __restrict__ pWed,
                                                int eBase, int lane, float* lds /*16*192*/)
{
    int arow = eBase + (lane & 15);
    const _Float16* Er = eh + (size_t)arow * EDGE_OUT;
    AFrag a0, a1;
    load_afrag_f16(Er, 0, lane, a0);
    load_afrag_f16(Er, 32, lane, a1);
    int cb = lane & 15;
#pragma unroll
    for (int tn = 0; tn < HC / 16; ++tn) {          // 12 column tiles
        CFrag acc;
#pragma unroll
        for (int i = 0; i < 8; ++i) acc.f[i] = 0.0f;
        v16h b0 = load_bfrag_packed(pWed, tn, 0, 2, lane);
        v16h b1 = load_bfrag_packed(pWed, tn, 1, 2, lane);
        acc.v = __builtin_amdgcn_wmma_f32_16x16x32_f16(
            false, a0.v, false, b0, (short)0, acc.v, false, false);
        acc.v = __builtin_amdgcn_wmma_f32_16x16x32_f16(
            false, a1.v, false, b1, (short)0, acc.v, false, false);
        int col = tn * 16 + cb;
#pragma unroll
        for (int r = 0; r < 8; ++r) {
            int row = r + ((lane & 16) ? 8 : 0);
            lds[row * HC + col] = acc.f[r];
        }
    }
}

// ---------------- pass 1: alpha + segment-max (ordered-uint atomicMax) ------
__global__ void attn_alpha_kernel(const _Float16* __restrict__ eh, const _Float16* __restrict__ pWed,
                                  const float* __restrict__ qbuf, const float* __restrict__ kbuf,
                                  const int* __restrict__ ei, float* __restrict__ alphaBuf,
                                  unsigned* __restrict__ amaxOrd)
{
    __shared__ __align__(16) float ldsEE[2][16 * HC];
    __shared__ int ldsIdx[2][32];
    int lane = threadIdx.x & 31;
    int wv   = threadIdx.x >> 5;
    int eBase = (blockIdx.x * 2 + wv) * 16;

    if (lane < 16) {
        ldsIdx[wv][lane]      = ei[eBase + lane];           // src
        ldsIdx[wv][16 + lane] = ei[EEDGE + eBase + lane];   // dst
    }
    compute_ee_tile(eh, pWed, eBase, lane, ldsEE[wv]);
    __syncthreads();

    int row = lane & 15, half = lane >> 4;
    int s = ldsIdx[wv][row], d = ldsIdx[wv][16 + row];
    const float* qd = qbuf + (size_t)d * HC;
    const float* ks = kbuf + (size_t)s * HC;
    const float* ee = &ldsEE[wv][row * HC];
    float p[NH] = {0.f, 0.f, 0.f};
#pragma unroll
    for (int h = 0; h < NH; ++h) {
        int base = h * NODE_OUT + half * 32;
#pragma unroll
        for (int c4 = 0; c4 < 8; ++c4) {
            float4 qv = *(const float4*)(qd + base + 4 * c4);
            float4 kv = *(const float4*)(ks + base + 4 * c4);
            float4 ev = *(const float4*)(ee + base + 4 * c4);
            p[h] += qv.x * (kv.x + ev.x) + qv.y * (kv.y + ev.y)
                  + qv.z * (kv.z + ev.z) + qv.w * (kv.w + ev.w);
        }
    }
#pragma unroll
    for (int h = 0; h < NH; ++h) {
        float tot = p[h] + __shfl_xor(p[h], 16);
        if (half == 0) {
            float al = tot * 0.125f;                        // / sqrt(64)
            alphaBuf[(size_t)(eBase + row) * NH + h] = al;
            atomicMax(&amaxOrd[d * NH + h], f2ord(al));
        }
    }
}

// ---------------- pass 2: ex = exp(alpha - amax); denom += ex ---------------
__global__ void attn_expsum_kernel(const int* __restrict__ ei, float* __restrict__ alphaBuf,
                                   const unsigned* __restrict__ amaxOrd, float* __restrict__ denom)
{
    int e = blockIdx.x * blockDim.x + threadIdx.x;
    if (e >= EEDGE) return;
    int d = ei[EEDGE + e];
#pragma unroll
    for (int h = 0; h < NH; ++h) {
        float a = alphaBuf[(size_t)e * NH + h];
        float m = ord2f(amaxOrd[d * NH + h]);
        float ex = __expf(a - m);
        alphaBuf[(size_t)e * NH + h] = ex;
        atomicAdd(&denom[d * NH + h], ex);
    }
}

// ---------------- pass 3: msg = (v[src]+ee) * w ; scatter-add to agg --------
__global__ void attn_agg_kernel(const _Float16* __restrict__ eh, const _Float16* __restrict__ pWed,
                                const float* __restrict__ vbuf, const int* __restrict__ ei,
                                const float* __restrict__ exBuf, const float* __restrict__ denom,
                                float* __restrict__ agg)
{
    __shared__ __align__(16) float ldsEE[2][16 * HC];
    __shared__ int   ldsIdx[2][32];
    __shared__ float ldsW[2][16 * NH];
    int lane = threadIdx.x & 31;
    int wv   = threadIdx.x >> 5;
    int eBase = (blockIdx.x * 2 + wv) * 16;

    if (lane < 16) {
        ldsIdx[wv][lane]      = ei[eBase + lane];
        ldsIdx[wv][16 + lane] = ei[EEDGE + eBase + lane];
    }
    compute_ee_tile(eh, pWed, eBase, lane, ldsEE[wv]);
    __syncthreads();
    if (lane < 16) {
        int d = ldsIdx[wv][16 + lane];
#pragma unroll
        for (int h = 0; h < NH; ++h)
            ldsW[wv][lane * NH + h] =
                exBuf[(size_t)(eBase + lane) * NH + h] / denom[d * NH + h];
    }
    __syncthreads();

    for (int r = 0; r < 16; ++r) {
        int s = ldsIdx[wv][r], d = ldsIdx[wv][16 + r];
        const float* vs = vbuf + (size_t)s * HC;
#pragma unroll
        for (int cc = 0; cc < HC / 32; ++cc) {
            int c = cc * 32 + lane;
            int h = c >> 6;
            float msg = (vs[c] + ldsEE[wv][r * HC + c]) * ldsW[wv][r * NH + h];
            atomicAdd(&agg[(size_t)d * HC + c], msg);
        }
    }
}

// ---------------- node_mlp_2: xg = relu(cat(feat, agg+skip, glob[b]) @ W + b)
__global__ void node2_kernel(const float* __restrict__ feat, const float* __restrict__ agg,
                             const float* __restrict__ skipb, const float* __restrict__ glob,
                             const int* __restrict__ batch, const _Float16* __restrict__ pW,
                             const float* __restrict__ bias, float* __restrict__ xg)
{
    const int K = NODE_OUT * 4 + GLOB_IN;           // 320
    const int chunks = K >> 5;                      // 10
    int lane = threadIdx.x & 31;
    int wave = blockIdx.x * (blockDim.x >> 5) + (threadIdx.x >> 5);
    const int tilesN = NODE_OUT >> 4;               // 4
    int tilesM = (NN + 15) >> 4;
    if (wave >= tilesM * tilesN) return;
    int tm = wave / tilesN, tn = wave % tilesN;

    int arow = tm * 16 + (lane & 15);
    int r = arow < NN ? arow : NN - 1;              // clamp; stores guarded
    const float* fr = feat  + (size_t)r * NODE_OUT;
    const float* ar = agg   + (size_t)r * HC;
    const float* sr = skipb + (size_t)r * HC;
    const float* gr = glob  + (size_t)batch[r] * GLOB_IN;
    int bcol = tn * 16 + (lane & 15);

    CFrag acc;
#pragma unroll
    for (int i = 0; i < 8; ++i) acc.f[i] = 0.0f;

#pragma unroll
    for (int kc = 0; kc < chunks; ++kc) {
        int kk = kc * 32;
        AFrag a;
        if (kk < NODE_OUT) {                         // feat region
            load_afrag_f32(fr, kk, lane, a);
        } else if (kk < NODE_OUT + HC) {             // agg + skip region
            const float* ra = ar + (kk - NODE_OUT) + ((lane & 16) ? 8 : 0);
            const float* rs = sr + (kk - NODE_OUT) + ((lane & 16) ? 8 : 0);
#pragma unroll
            for (int p = 0; p < 8; ++p) {
                int kb = (p < 4) ? 2 * p : 16 + 2 * (p - 4);
                float2 va = *(const float2*)(ra + kb);
                float2 vs = *(const float2*)(rs + kb);
                a.h[2 * p]     = (_Float16)(va.x + vs.x);
                a.h[2 * p + 1] = (_Float16)(va.y + vs.y);
            }
        } else {                                     // glob[batch] region
            load_afrag_f32(gr, kk - NODE_OUT - HC, lane, a);
        }
        v16h b = load_bfrag_packed(pW, tn, kc, chunks, lane);
        acc.v = __builtin_amdgcn_wmma_f32_16x16x32_f16(
            false, a.v, false, b, (short)0, acc.v, false, false);
    }

    float bv = bias[bcol];
#pragma unroll
    for (int e = 0; e < 8; ++e) {
        int row = tm * 16 + e + ((lane & 16) ? 8 : 0);
        if (row < NN)
            xg[(size_t)row * NODE_OUT + bcol] = fmaxf(acc.f[e] + bv, 0.0f);
    }
}

// ---------------- graph scatter-mean accumulation ---------------------------
__global__ void graph_sum_kernel(const float* __restrict__ xg, const int* __restrict__ batch,
                                 float* __restrict__ sums, float* __restrict__ counts)
{
    int i = blockIdx.x * blockDim.x + threadIdx.x;
    if (i >= NN * NODE_OUT) return;
    int n = i >> 6, c = i & 63;
    int b = batch[n];
    atomicAdd(&sums[b * NODE_OUT + c], xg[i]);
    if (c == 0) atomicAdd(&counts[b], 1.0f);
}

// ---------------- global MLP: u = relu(cat(glob, mean) @ Wg + b) ------------
__global__ void global_mlp_kernel(const float* __restrict__ glob, const float* __restrict__ sums,
                                  const float* __restrict__ counts, const _Float16* __restrict__ pW,
                                  const float* __restrict__ bias, float* __restrict__ u)
{
    const int K = GLOB_IN + NODE_OUT;               // 128
    const int chunks = K >> 5;                      // 4
    int lane = threadIdx.x & 31;
    int wv   = threadIdx.x >> 5;                    // 8 waves: 2x4 tiles
    int tm = wv >> 2, tn = wv & 3;

    int arow = tm * 16 + (lane & 15);               // < 32 always
    int bcol = tn * 16 + (lane & 15);
    float invc = 1.0f / fmaxf(counts[arow], 1.0f);
    const float* gr = glob + (size_t)arow * GLOB_IN;
    const float* sr = sums + (size_t)arow * NODE_OUT;

    CFrag acc;
#pragma unroll
    for (int i = 0; i < 8; ++i) acc.f[i] = 0.0f;

#pragma unroll
    for (int kc = 0; kc < chunks; ++kc) {
        int kk = kc * 32;
        AFrag a;
        if (kk < GLOB_IN) {
            load_afrag_f32(gr, kk, lane, a);
        } else {
            const float* rr = sr + (kk - GLOB_IN) + ((lane & 16) ? 8 : 0);
#pragma unroll
            for (int p = 0; p < 8; ++p) {
                int kb = (p < 4) ? 2 * p : 16 + 2 * (p - 4);
                float2 v = *(const float2*)(rr + kb);
                a.h[2 * p]     = (_Float16)(v.x * invc);
                a.h[2 * p + 1] = (_Float16)(v.y * invc);
            }
        }
        v16h b = load_bfrag_packed(pW, tn, kc, chunks, lane);
        acc.v = __builtin_amdgcn_wmma_f32_16x16x32_f16(
            false, a.v, false, b, (short)0, acc.v, false, false);
    }

    float bv = bias[bcol];
#pragma unroll
    for (int e = 0; e < 8; ++e) {
        int row = tm * 16 + e + ((lane & 16) ? 8 : 0);
        u[(size_t)row * GLOB_OUT + bcol] = fmaxf(acc.f[e] + bv, 0.0f);
    }
}

// ---------------- host-side orchestration ----------------------------------
extern "C" void kernel_launch(void* const* d_in, const int* in_sizes, int n_in,
                              void* d_out, int out_size, void* d_ws, size_t ws_size,
                              hipStream_t stream) {
    const float* x        = (const float*)d_in[0];
    const float* edgeattr = (const float*)d_in[1];
    const float* glob     = (const float*)d_in[2];
    const int*   ei       = (const int*)  d_in[3];   // [2,E]: src row, dst row
    const int*   batch    = (const int*)  d_in[4];
    const float* We_w     = (const float*)d_in[5];
    const float* We_b     = (const float*)d_in[6];
    const float* Wn_w     = (const float*)d_in[7];
    const float* Wn_b     = (const float*)d_in[8];
    const float* Wq_w     = (const float*)d_in[9];
    const float* Wq_b     = (const float*)d_in[10];
    const float* Wk_w     = (const float*)d_in[11];
    const float* Wk_b     = (const float*)d_in[12];
    const float* Wv_w     = (const float*)d_in[13];
    const float* Wv_b     = (const float*)d_in[14];
    const float* Wed_w    = (const float*)d_in[15];
    const float* Wskip_w  = (const float*)d_in[16];
    const float* Wskip_b  = (const float*)d_in[17];
    const float* Wn2_w    = (const float*)d_in[18];
    const float* Wn2_b    = (const float*)d_in[19];
    const float* Wg_w     = (const float*)d_in[20];
    const float* Wg_b     = (const float*)d_in[21];

    // outputs: xg [N,64] | e [E,64] | u [B,64]
    float* xg_out = (float*)d_out;
    float* e_out  = xg_out + (size_t)NN * NODE_OUT;
    float* u_out  = e_out + (size_t)EEDGE * EDGE_OUT;

    // ---- workspace: f32 region ----
    float* ws = (float*)d_ws;
    size_t off = 0;
    float* qbuf   = ws + off; off += (size_t)NN * HC;
    float* kbuf   = ws + off; off += (size_t)NN * HC;
    float* vbuf   = ws + off; off += (size_t)NN * HC;
    float* skipb  = ws + off; off += (size_t)NN * HC;
    float* featb  = ws + off; off += (size_t)NN * NODE_OUT;
    float* alphaB = ws + off; off += (size_t)EEDGE * NH;
    float*    zeroBase = ws + off;
    unsigned* amaxOrd  = (unsigned*)(ws + off); off += (size_t)NN * NH;
    float*    denom    = ws + off;              off += (size_t)NN * NH;
    float*    agg      = ws + off;              off += (size_t)NN * HC;
    float*    sums     = ws + off;              off += (size_t)BG * NODE_OUT;
    float*    counts   = ws + off;              off += (size_t)BG;
    long zeroCount = (long)((ws + off) - zeroBase);

    // ---- workspace: f16 region (64B aligned for v16h fragment loads) ----
    size_t byteOff = (off * sizeof(float) + 63) & ~(size_t)63;
    _Float16* hb = (_Float16*)((char*)d_ws + byteOff);
    size_t ho = 0;
    _Float16* xh    = hb + ho; ho += (size_t)NN * NODE_IN;
    _Float16* eh    = hb + ho; ho += (size_t)EEDGE * EDGE_OUT;
    _Float16* pWe   = hb + ho; ho += (NODE_IN + EDGE_IN) * EDGE_OUT;
    _Float16* pWn   = hb + ho; ho += NODE_IN * NODE_OUT;
    _Float16* pWq   = hb + ho; ho += NODE_IN * HC;
    _Float16* pWk   = hb + ho; ho += NODE_IN * HC;
    _Float16* pWv   = hb + ho; ho += NODE_IN * HC;
    _Float16* pWsk  = hb + ho; ho += NODE_IN * HC;
    _Float16* pWed  = hb + ho; ho += EDGE_OUT * HC;
    _Float16* pWn2  = hb + ho; ho += (NODE_OUT * 4 + GLOB_IN) * NODE_OUT;
    _Float16* pWg   = hb + ho; ho += (GLOB_IN + NODE_OUT) * GLOB_OUT;

    // 0) init + conversions + weight packing
    zero_kernel<<<2048, 256, 0, stream>>>(zeroBase, zeroCount);
    cvt_f16_kernel<<<(NN * NODE_IN + 255) / 256, 256, 0, stream>>>(x, xh, (long)NN * NODE_IN);
    auto pack = [&](const float* W, _Float16* P, int K, int Ncols) {
        pack_weight_kernel<<<(K * Ncols + 255) / 256, 256, 0, stream>>>(W, P, K, Ncols);
    };
    pack(We_w,    pWe,  NODE_IN + EDGE_IN, EDGE_OUT);
    pack(Wn_w,    pWn,  NODE_IN, NODE_OUT);
    pack(Wq_w,    pWq,  NODE_IN, HC);
    pack(Wk_w,    pWk,  NODE_IN, HC);
    pack(Wv_w,    pWv,  NODE_IN, HC);
    pack(Wskip_w, pWsk, NODE_IN, HC);
    pack(Wed_w,   pWed, EDGE_OUT, HC);
    pack(Wn2_w,   pWn2, NODE_OUT * 4 + GLOB_IN, NODE_OUT);
    pack(Wg_w,    pWg,  GLOB_IN + NODE_OUT, GLOB_OUT);

    // 1) node projections
    {
        int tiles = ((NN + 15) / 16) * (HC / 16);
        int blocks = (tiles + 7) / 8;
        wmma_gemm_f16<<<blocks, 256, 0, stream>>>(xh, pWq,  Wq_b,    qbuf,  NN, NODE_IN, HC, 0);
        wmma_gemm_f16<<<blocks, 256, 0, stream>>>(xh, pWk,  Wk_b,    kbuf,  NN, NODE_IN, HC, 0);
        wmma_gemm_f16<<<blocks, 256, 0, stream>>>(xh, pWv,  Wv_b,    vbuf,  NN, NODE_IN, HC, 0);
        wmma_gemm_f16<<<blocks, 256, 0, stream>>>(xh, pWsk, Wskip_b, skipb, NN, NODE_IN, HC, 0);
        int tilesF = ((NN + 15) / 16) * (NODE_OUT / 16);
        wmma_gemm_f16<<<(tilesF + 7) / 8, 256, 0, stream>>>(xh, pWn, Wn_b, featb, NN, NODE_IN, NODE_OUT, 0);
    }

    // 2) edge MLP -> e (f32 into d_out, f16 copy for attention passes)
    {
        int tiles = (EEDGE / 16) * (EDGE_OUT / 16);
        edge_mlp_kernel<<<(tiles + 7) / 8, 256, 0, stream>>>(xh, edgeattr, ei, pWe, We_b, e_out, eh);
    }

    // 3) alpha + segment max (ee recomputed via WMMA, never materialized)
    attn_alpha_kernel<<<EEDGE / 32, 64, 0, stream>>>(eh, pWed, qbuf, kbuf, ei, alphaB, amaxOrd);

    // 4) exp + denominator
    attn_expsum_kernel<<<(EEDGE + 255) / 256, 256, 0, stream>>>(ei, alphaB, amaxOrd, denom);

    // 5) weighted message aggregation (ee recomputed again; scatter-add)
    attn_agg_kernel<<<EEDGE / 32, 64, 0, stream>>>(eh, pWed, vbuf, ei, alphaB, denom, agg);

    // 6) node_mlp_2 -> xg
    {
        int tiles = ((NN + 15) / 16) * (NODE_OUT / 16);
        node2_kernel<<<(tiles + 7) / 8, 256, 0, stream>>>(featb, agg, skipb, glob, batch,
                                                          pWn2, Wn2_b, xg_out);
    }

    // 7) per-graph mean accumulation
    graph_sum_kernel<<<(NN * NODE_OUT + 255) / 256, 256, 0, stream>>>(xg_out, batch, sums, counts);

    // 8) global MLP -> u
    global_mlp_kernel<<<1, 256, 0, stream>>>(glob, sums, counts, pWg, Wg_b, u_out);
}
